// GPT_9131100471754
// MI455X (gfx1250) — matrix operand order repaired and checked by reading, not compiled, not run on hardware
//
#include <hip/hip_runtime.h>
#include <hip/hip_bf16.h>

// ---------------------------------------------------------------------------
// GPT forward for MI455X (gfx1250, wave32, WMMA 16x16x32 f16 -> f32 acc,
// async global->LDS staging when the toolchain exposes the builtins)
// ---------------------------------------------------------------------------

typedef __attribute__((ext_vector_type(16))) _Float16 v16h;
typedef __attribute__((ext_vector_type(8)))  float    v8f;
typedef int gpt_v4i __attribute__((vector_size(16)));

#define GPT_L   6
#define GPT_H   8
#define GPT_D   512
#define GPT_V   50304
#define GPT_T   2048
#define GPT_DFF 2048
#define GPT_DH  64

// ---- async global->LDS (CDNA5, ASYNCcnt) -----------------------------------
#if defined(__has_builtin)
#if __has_builtin(__builtin_amdgcn_global_load_async_to_lds_b128)
#define HAVE_ASYNC 1
#endif
#endif
#ifndef HAVE_ASYNC
#define HAVE_ASYNC 0
#endif

#if HAVE_ASYNC
#if __has_builtin(__builtin_amdgcn_s_wait_asynccnt)
#define WAIT_ASYNC(n) __builtin_amdgcn_s_wait_asynccnt(n)
#else
#define WAIT_ASYNC(n) asm volatile("s_wait_asynccnt %0" ::"n"(n) : "memory")
#endif
// copy 16 bytes (8 f16) global -> LDS per lane, tracked by ASYNCcnt
#define GASYNC16(ldst, gsrc)                                                   \
  __builtin_amdgcn_global_load_async_to_lds_b128(                              \
      (__attribute__((address_space(1))) gpt_v4i*)(gsrc),                      \
      (__attribute__((address_space(3))) gpt_v4i*)(ldst), 0, 0)
#else
#define WAIT_ASYNC(n) ((void)0)
#endif

// ---- WMMA fragment loaders (contiguous k = 16*(l>=16)+j for A and B, wave32) ----------
// A 16x32 f16: lane l row M=l&15; halves j=0..15 -> K=hi+j
// (hi = 16*(l>=16)): one contiguous 32B run. `ld` multiple of 8, base 16B.
__device__ __forceinline__ v16h load_a16x32_vec(const _Float16* base, int ld) {
  const int l = threadIdx.x & 31;
  const _Float16* row = base + (l & 15) * ld + ((l >> 4) << 4);
  union { v16h v; uint4 q[2]; } u;
  u.q[0] = *(const uint4*)(row);
  u.q[1] = *(const uint4*)(row + 8);
  return u.v;
}

// B 32x16 f16 from a TRANSPOSED [N][K] tile: lane l col N=l&15; halves j ->
// K = khi + j (khi = 16*(l>=16)): one contiguous 32B run.
__device__ __forceinline__ v16h load_bT_vec(const _Float16* baseT, int ld) {
  const int l = threadIdx.x & 31;
  const _Float16* row = baseT + (l & 15) * ld + ((l >> 4) << 4);
  union { v16h v; uint4 q[2]; } u;
  u.q[0] = *(const uint4*)(row);
  u.q[1] = *(const uint4*)(row + 8);
  return u.v;
}

// B 32x16 f16 from row-major [K x N] (scalar gather; attention P@V only)
__device__ __forceinline__ v16h load_b32x16(const _Float16* base, int ld) {
  const int l   = threadIdx.x & 31;
  const int n   = l & 15;
  const int khi = (l >> 4) << 4;
  v16h b;
#pragma unroll
  for (int j = 0; j < 16; ++j) b[j] = base[(khi + j) * ld + n];
  return b;
}

__device__ __forceinline__ float gelu_tanh(float x) {
  const float c = 0.7978845608028654f;
  return 0.5f * x * (1.0f + tanhf(c * (x + 0.044715f * x * x * x)));
}

// ---------------------------------------------------------------------------
// WMMA GEMM: C[M,N] = act(A[M,K] @ B[K,N] + bias) (+ residual), B given
// PRE-TRANSPOSED as BT[N][K]. Block 256 thr = 8 waves; block tile 128x64;
// wave tile 32x32; K-step 32; double-buffered async global->LDS staging.
// ---------------------------------------------------------------------------
#define BM 128
#define BN 64
#define BK 32
#define ASTR 40   // padded LDS stride (halves): 20*m mod 64 is a permutation
#define BSTR 40

__global__ __launch_bounds__(256) void gemm_f16_kernel(
    const _Float16* __restrict__ A, const _Float16* __restrict__ BT,
    const float* __restrict__ bias, const float* __restrict__ resid,
    float* __restrict__ Cf, _Float16* __restrict__ Ch,
    int M, int N, int K, int act)
{
  __shared__ __align__(16) _Float16 As[2][BM * ASTR];
  __shared__ __align__(16) _Float16 BsT[2][BN * BSTR];

  const int tid  = threadIdx.x;
  const int wave = tid >> 5;
  const int wr   = wave >> 1;  // 0..3 : 32-row slice
  const int wc   = wave & 1;   // 0..1 : 32-col slice
  const int m0   = blockIdx.y * BM;
  const int n0   = blockIdx.x * BN;

  v8f acc00 = {}, acc01 = {}, acc10 = {}, acc11 = {};

  // staging maps: A 128x32 (16 halves/thread), BT 64x32 (8 halves/thread)
  const int arow = tid >> 1;            // 0..127
  const int acol = (tid & 1) << 4;      // 0,16
  const int bn   = tid >> 2;            // 0..63
  const int bkc  = (tid & 3) << 3;      // 0,8,16,24

  const _Float16* aSrcBase = A + (size_t)(m0 + arow) * K + acol;
  const _Float16* bSrcBase = BT + (size_t)(n0 + bn) * K + bkc;
  _Float16* aDst0 = &As[0][arow * ASTR + acol];
  _Float16* bDst0 = &BsT[0][bn * BSTR + bkc];

#if HAVE_ASYNC
#define STAGE_CP(d, s) GASYNC16(d, s)
#else
#define STAGE_CP(d, s) (*(uint4*)(d) = *(const uint4*)(s))
#endif

#define STAGE_TILE(buf, kk)                                                    \
  do {                                                                         \
    const _Float16* as_ = aSrcBase + (kk);                                     \
    const _Float16* bs_ = bSrcBase + (kk);                                     \
    _Float16* ad_ = aDst0 + (buf) * (BM * ASTR);                               \
    _Float16* bd_ = bDst0 + (buf) * (BN * BSTR);                               \
    STAGE_CP(ad_, as_);                                                        \
    STAGE_CP(ad_ + 8, as_ + 8);                                                \
    STAGE_CP(bd_, bs_);                                                        \
  } while (0)

  STAGE_TILE(0, 0);

  int buf = 0;
  for (int k0 = 0; k0 < K; k0 += BK, buf ^= 1) {
    const bool more = (k0 + BK) < K;
    if (more) {
      STAGE_TILE(buf ^ 1, k0 + BK);
      WAIT_ASYNC(3);   // each thread: 3 in-flight b128 per tile, in-order done
    } else {
      WAIT_ASYNC(0);
    }
    __syncthreads();

    const _Float16* as = As[buf];
    const _Float16* bs = BsT[buf];
    const v16h a0 = load_a16x32_vec(as + (wr * 32) * ASTR, ASTR);
    const v16h a1 = load_a16x32_vec(as + (wr * 32 + 16) * ASTR, ASTR);
    const v16h b0 = load_bT_vec(bs + (wc * 32) * BSTR, BSTR);
    const v16h b1 = load_bT_vec(bs + (wc * 32 + 16) * BSTR, BSTR);

    acc00 = __builtin_amdgcn_wmma_f32_16x16x32_f16(false, a0, false, b0,
                                                   (short)0, acc00, false, false);
    acc01 = __builtin_amdgcn_wmma_f32_16x16x32_f16(false, a0, false, b1,
                                                   (short)0, acc01, false, false);
    acc10 = __builtin_amdgcn_wmma_f32_16x16x32_f16(false, a1, false, b0,
                                                   (short)0, acc10, false, false);
    acc11 = __builtin_amdgcn_wmma_f32_16x16x32_f16(false, a1, false, b1,
                                                   (short)0, acc11, false, false);
    __syncthreads();
  }

  const int l     = tid & 31;
  const int mbase = m0 + wr * 32 + ((l >> 4) << 3);
  const int nbase = n0 + wc * 32 + (l & 15);
#pragma unroll
  for (int q = 0; q < 4; ++q) {
    const v8f* acc = (q == 0) ? &acc00 : (q == 1) ? &acc01 : (q == 2) ? &acc10 : &acc11;
    const int rofs = (q >= 2) ? 16 : 0;
    const int cofs = (q & 1) ? 16 : 0;
#pragma unroll
    for (int v = 0; v < 8; ++v) {
      const int row = mbase + rofs + v;
      const int col = nbase + cofs;
      float val = (*acc)[v];
      if (bias)  val += bias[col];
      if (act)   val  = gelu_tanh(val);
      if (resid) val += resid[(size_t)row * N + col];
      if (Cf) Cf[(size_t)row * N + col] = val;
      else    Ch[(size_t)row * N + col] = (_Float16)val;
    }
  }
}

// ---------------------------------------------------------------------------
// Fused causal flash-attention. One wave32 per (16-query tile, head).
// ---------------------------------------------------------------------------
#define KVSTR 72   // padded K/V LDS stride (halves): 36*n mod 64 permutation
#define PSTR  40

__global__ __launch_bounds__(32) void attn_kernel(
    const _Float16* __restrict__ Q, const _Float16* __restrict__ Km,
    const _Float16* __restrict__ Vm, _Float16* __restrict__ O)
{
  __shared__ __align__(16) _Float16 Kt[32 * KVSTR];
  __shared__ __align__(16) _Float16 Vt[32 * KVSTR];
  __shared__ __align__(16) _Float16 Pt[16 * PSTR];

  const int l     = threadIdx.x;
  const int qt    = blockIdx.x;
  const int h     = blockIdx.y;
  const int qrow0 = qt * 16;
  const int hoff  = h * GPT_DH;
  const float scale = 0.125f; // 1/sqrt(64)

  const v16h a0 = load_a16x32_vec(Q + (size_t)qrow0 * GPT_D + hoff, GPT_D);
  const v16h a1 = load_a16x32_vec(Q + (size_t)qrow0 * GPT_D + hoff + 32, GPT_D);

  v8f o0 = {}, o1 = {}, o2 = {}, o3 = {};
  float rm[8], rs[8];
#pragma unroll
  for (int v = 0; v < 8; ++v) { rm[v] = -3.0e38f; rs[v] = 0.0f; }

  const int mhalf = (l >> 4) << 3;
  const int nl    = l & 15;

  const int nit = (qrow0 + 16 + 31) >> 5;
  for (int it = 0; it < nit; ++it) {
    const int kr0 = it << 5;

    // stage 32 K rows + 32 V rows (64 f16 each); lane owns one row of each
    {
      const _Float16* ks = Km + (size_t)(kr0 + l) * GPT_D + hoff;
      const _Float16* vs = Vm + (size_t)(kr0 + l) * GPT_D + hoff;
      _Float16* kd = Kt + l * KVSTR;
      _Float16* vd = Vt + l * KVSTR;
#if HAVE_ASYNC
#pragma unroll
      for (int j = 0; j < 8; ++j) {
        GASYNC16(kd + j * 8, ks + j * 8);
        GASYNC16(vd + j * 8, vs + j * 8);
      }
      WAIT_ASYNC(0);
#else
#pragma unroll
      for (int j = 0; j < 8; ++j) {
        ((uint4*)kd)[j] = ((const uint4*)ks)[j];
        ((uint4*)vd)[j] = ((const uint4*)vs)[j];
      }
#endif
    }
    __syncthreads();

    // S = Q K^T : B frag k-dim = head dim (contiguous in Kt rows) -> b128
    v8f s0 = {}, s1 = {};
    {
      const _Float16* kb0 = Kt + (nl) * KVSTR + ((l >> 4) << 4);
      const _Float16* kb1 = Kt + (16 + nl) * KVSTR + ((l >> 4) << 4);
      union { v16h v; uint4 q[2]; } u;
      u.q[0] = *(const uint4*)(kb0);      u.q[1] = *(const uint4*)(kb0 + 8);
      s0 = __builtin_amdgcn_wmma_f32_16x16x32_f16(false, a0, false, u.v,
                                                  (short)0, s0, false, false);
      u.q[0] = *(const uint4*)(kb0 + 32); u.q[1] = *(const uint4*)(kb0 + 40);
      s0 = __builtin_amdgcn_wmma_f32_16x16x32_f16(false, a1, false, u.v,
                                                  (short)0, s0, false, false);
      u.q[0] = *(const uint4*)(kb1);      u.q[1] = *(const uint4*)(kb1 + 8);
      s1 = __builtin_amdgcn_wmma_f32_16x16x32_f16(false, a0, false, u.v,
                                                  (short)0, s1, false, false);
      u.q[0] = *(const uint4*)(kb1 + 32); u.q[1] = *(const uint4*)(kb1 + 40);
      s1 = __builtin_amdgcn_wmma_f32_16x16x32_f16(false, a1, false, u.v,
                                                  (short)0, s1, false, false);
    }

    // scale + causal mask + online softmax per C row (v, half)
#pragma unroll
    for (int v = 0; v < 8; ++v) {
      const int qg = qrow0 + mhalf + v;
      float x0 = (kr0 + nl      <= qg) ? s0[v] * scale : -3.0e38f;
      float x1 = (kr0 + 16 + nl <= qg) ? s1[v] * scale : -3.0e38f;

      float t = fmaxf(x0, x1);
      t = fmaxf(t, __shfl_xor(t, 1, 32));
      t = fmaxf(t, __shfl_xor(t, 2, 32));
      t = fmaxf(t, __shfl_xor(t, 4, 32));
      t = fmaxf(t, __shfl_xor(t, 8, 32));

      const float nm = fmaxf(rm[v], t);
      const float f  = __expf(rm[v] - nm);
      rm[v] = nm;

      const float p0 = __expf(x0 - nm);
      const float p1 = __expf(x1 - nm);
      float ps = p0 + p1;
      ps += __shfl_xor(ps, 1, 32);
      ps += __shfl_xor(ps, 2, 32);
      ps += __shfl_xor(ps, 4, 32);
      ps += __shfl_xor(ps, 8, 32);
      rs[v] = rs[v] * f + ps;

      o0[v] *= f; o1[v] *= f; o2[v] *= f; o3[v] *= f;

      const int m = mhalf + v;
      Pt[m * PSTR + nl]      = (_Float16)p0;
      Pt[m * PSTR + 16 + nl] = (_Float16)p1;
    }
    __syncthreads();

    // O += P @ V
    const v16h pa = load_a16x32_vec(Pt, PSTR);
    o0 = __builtin_amdgcn_wmma_f32_16x16x32_f16(false, pa, false,
           load_b32x16(Vt + 0,  KVSTR), (short)0, o0, false, false);
    o1 = __builtin_amdgcn_wmma_f32_16x16x32_f16(false, pa, false,
           load_b32x16(Vt + 16, KVSTR), (short)0, o1, false, false);
    o2 = __builtin_amdgcn_wmma_f32_16x16x32_f16(false, pa, false,
           load_b32x16(Vt + 32, KVSTR), (short)0, o2, false, false);
    o3 = __builtin_amdgcn_wmma_f32_16x16x32_f16(false, pa, false,
           load_b32x16(Vt + 48, KVSTR), (short)0, o3, false, false);
    __syncthreads();
  }

  // normalize and store f16 [T,D]
#pragma unroll
  for (int v = 0; v < 8; ++v) {
    const float inv = 1.0f / rs[v];
    const int row = qrow0 + mhalf + v;
    _Float16* dst = O + (size_t)row * GPT_D + hoff + nl;
    dst[0]  = (_Float16)(o0[v] * inv);
    dst[16] = (_Float16)(o1[v] * inv);
    dst[32] = (_Float16)(o2[v] * inv);
    dst[48] = (_Float16)(o3[v] * inv);
  }
}

// ---------------------------------------------------------------------------
// LayerNorm over D=512, one 256-thread block per row, f16 output
// ---------------------------------------------------------------------------
__global__ __launch_bounds__(256) void ln_kernel(
    const float* __restrict__ x, const float* __restrict__ scale,
    const float* __restrict__ bias, _Float16* __restrict__ out)
{
  __shared__ float red[256];
  const int row = blockIdx.x;
  const int t   = threadIdx.x;
  const float* xr = x + (size_t)row * GPT_D;
  const float x0 = xr[t], x1 = xr[t + 256];

  red[t] = x0 + x1;
  __syncthreads();
  for (int o = 128; o > 0; o >>= 1) { if (t < o) red[t] += red[t + o]; __syncthreads(); }
  const float mean = red[0] * (1.0f / GPT_D);
  __syncthreads();

  const float d0 = x0 - mean, d1 = x1 - mean;
  red[t] = d0 * d0 + d1 * d1;
  __syncthreads();
  for (int o = 128; o > 0; o >>= 1) { if (t < o) red[t] += red[t + o]; __syncthreads(); }
  const float r = rsqrtf(red[0] * (1.0f / GPT_D) + 1e-6f);

  _Float16* orow = out + (size_t)row * GPT_D;
  orow[t]       = (_Float16)(d0 * r * scale[t]       + bias[t]);
  orow[t + 256] = (_Float16)(d1 * r * scale[t + 256] + bias[t + 256]);
}

// ---------------------------------------------------------------------------
// Embedding gather + positional add
// ---------------------------------------------------------------------------
__global__ void embed_kernel(const int* __restrict__ ids,
                             const float* __restrict__ tok,
                             const float* __restrict__ pos,
                             float* __restrict__ x)
{
  const int t  = blockIdx.x;
  const int id = ids[t];
  for (int i = threadIdx.x; i < GPT_D; i += blockDim.x)
    x[(size_t)t * GPT_D + i] = tok[(size_t)id * GPT_D + i] + pos[(size_t)t * GPT_D + i];
}

// ---------------------------------------------------------------------------
// fp32 -> f16 conversion (n multiple of 1024)
// ---------------------------------------------------------------------------
__global__ void cvt_f32_f16_kernel(const float* __restrict__ src,
                                   _Float16* __restrict__ dst, int n)
{
  const int i = (blockIdx.x * blockDim.x + threadIdx.x) * 4;
  if (i < n) {
    const float4 f = *(const float4*)(src + i);
    dst[i]     = (_Float16)f.x;
    dst[i + 1] = (_Float16)f.y;
    dst[i + 2] = (_Float16)f.z;
    dst[i + 3] = (_Float16)f.w;
  }
}

// ---------------------------------------------------------------------------
// fp32 [K][N] -> f16 TRANSPOSED [N][K]; 64x64 tiles through padded LDS.
// Both global read and write fully coalesced (16B per thread per op).
// ---------------------------------------------------------------------------
__global__ __launch_bounds__(256) void cvtT_kernel(
    const float* __restrict__ src, _Float16* __restrict__ dst, int K, int N)
{
  __shared__ __align__(16) _Float16 tile[64][72];
  const int t  = threadIdx.x;
  const int k0 = blockIdx.y * 64;
  const int n0 = blockIdx.x * 64;
  const int r  = t >> 2;          // 0..63
  const int c  = (t & 3) << 4;    // 0,16,32,48

  const float4* s = (const float4*)(src + (size_t)(k0 + r) * N + n0 + c);
#pragma unroll
  for (int q = 0; q < 4; ++q) {
    const float4 f = s[q];
    _Float16* d = &tile[r][c + q * 4];
    d[0] = (_Float16)f.x; d[1] = (_Float16)f.y;
    d[2] = (_Float16)f.z; d[3] = (_Float16)f.w;
  }
  __syncthreads();

  union { uint4 q[2]; _Float16 h[16]; } u;
#pragma unroll
  for (int j = 0; j < 16; ++j) u.h[j] = tile[c + j][r];
  uint4* d = (uint4*)(dst + (size_t)(n0 + r) * K + k0 + c);
  d[0] = u.q[0];
  d[1] = u.q[1];
}

// ---------------------------------------------------------------------------
// host driver
// ---------------------------------------------------------------------------
extern "C" void kernel_launch(void* const* d_in, const int* in_sizes, int n_in,
                              void* d_out, int out_size, void* d_ws, size_t ws_size,
                              hipStream_t stream) {
  (void)in_sizes; (void)n_in; (void)out_size; (void)ws_size;
  constexpr int L = GPT_L, D = GPT_D, V = GPT_V, T = GPT_T, DFF = GPT_DFF;
  constexpr size_t DD = (size_t)D * D;

  const int*   ids  = (const int*)  d_in[0];
  const float* tok  = (const float*)d_in[1];
  const float* pos  = (const float*)d_in[2];
  const float* l1s  = (const float*)d_in[3];
  const float* l1b  = (const float*)d_in[4];
  const float* wq   = (const float*)d_in[5];
  const float* bq   = (const float*)d_in[6];
  const float* wk   = (const float*)d_in[7];
  const float* bk   = (const float*)d_in[8];
  const float* wv   = (const float*)d_in[9];
  const float* bv   = (const float*)d_in[10];
  const float* wo   = (const float*)d_in[11];
  const float* bo   = (const float*)d_in[12];
  const float* l2s  = (const float*)d_in[13];
  const float* l2b  = (const float*)d_in[14];
  const float* w1   = (const float*)d_in[15];
  const float* b1   = (const float*)d_in[16];
  const float* w2   = (const float*)d_in[17];
  const float* b2   = (const float*)d_in[18];
  const float* lm   = (const float*)d_in[19];

  char* ws = (char*)d_ws;
  size_t off = 0;
  auto alloc = [&](size_t bytes) -> void* {
    void* p = ws + off;
    off += (bytes + 255) & ~(size_t)255;
    return p;
  };

  // transposed f16 weights: WT[N][K]
  _Float16* wq16 = (_Float16*)alloc(L * DD * 2);
  _Float16* wk16 = (_Float16*)alloc(L * DD * 2);
  _Float16* wv16 = (_Float16*)alloc(L * DD * 2);
  _Float16* wo16 = (_Float16*)alloc(L * DD * 2);
  _Float16* w116 = (_Float16*)alloc((size_t)L * D * DFF * 2);
  _Float16* w216 = (_Float16*)alloc((size_t)L * DFF * D * 2);
  _Float16* lm16 = (_Float16*)alloc((size_t)D * V * 2);
  float*    x    = (float*)   alloc((size_t)T * D * 4);
  _Float16* h16  = (_Float16*)alloc((size_t)T * D * 2);
  _Float16* q16  = (_Float16*)alloc((size_t)T * D * 2);
  _Float16* k16  = (_Float16*)alloc((size_t)T * D * 2);
  _Float16* v16b = (_Float16*)alloc((size_t)T * D * 2);
  _Float16* ao16 = (_Float16*)alloc((size_t)T * D * 2);
  _Float16* mid  = (_Float16*)alloc((size_t)T * DFF * 2);

  auto cvtT = [&](const float* s, _Float16* d, int K, int N) {
    cvtT_kernel<<<dim3(N / 64, K / 64), 256, 0, stream>>>(s, d, K, N);
  };
  for (int l = 0; l < L; ++l) {
    cvtT(wq + l * DD, wq16 + l * DD, D, D);
    cvtT(wk + l * DD, wk16 + l * DD, D, D);
    cvtT(wv + l * DD, wv16 + l * DD, D, D);
    cvtT(wo + l * DD, wo16 + l * DD, D, D);
    cvtT(w1 + (size_t)l * D * DFF, w116 + (size_t)l * D * DFF, D, DFF);
    cvtT(w2 + (size_t)l * DFF * D, w216 + (size_t)l * DFF * D, DFF, D);
  }
  cvtT(lm, lm16, D, V);

  embed_kernel<<<dim3(T), 256, 0, stream>>>(ids, tok, pos, x);

  auto gemm = [&](const _Float16* A, const _Float16* BT, const float* bias,
                  const float* resid, float* Cf, _Float16* Ch,
                  int M, int N, int K, int act) {
    gemm_f16_kernel<<<dim3(N / BN, M / BM), 256, 0, stream>>>(
        A, BT, bias, resid, Cf, Ch, M, N, K, act);
  };

  for (int l = 0; l < L; ++l) {
    ln_kernel<<<dim3(T), 256, 0, stream>>>(x, l1s + l * D, l1b + l * D, h16);

    gemm(h16, wq16 + l * DD, bq + l * D, nullptr, nullptr, q16, T, D, D, 0);
    gemm(h16, wk16 + l * DD, bk + l * D, nullptr, nullptr, k16, T, D, D, 0);
    gemm(h16, wv16 + l * DD, bv + l * D, nullptr, nullptr, v16b, T, D, D, 0);

    attn_kernel<<<dim3(T / 16, GPT_H), 32, 0, stream>>>(q16, k16, v16b, ao16);

    gemm(ao16, wo16 + l * DD, bo + l * D, x, x, nullptr, T, D, D, 0);

    ln_kernel<<<dim3(T), 256, 0, stream>>>(x, l2s + l * D, l2b + l * D, h16);

    gemm(h16, w116 + (size_t)l * D * DFF, b1 + l * DFF,
         nullptr, nullptr, mid, T, DFF, D, 1);
    gemm(mid, w216 + (size_t)l * DFF * D, b2 + l * D, x, x, nullptr, T, D, DFF, 0);
  }

  cvt_f32_f16_kernel<<<dim3((T * D) / 1024), 256, 0, stream>>>(x, h16, T * D);
  gemm(h16, lm16, nullptr, nullptr, (float*)d_out, nullptr, T, V, D, 0);
}
